// IterativeRefiner_5935644803520
// MI455X (gfx1250) — matrix operand order repaired and checked by reading, not compiled, not run on hardware
//
#include <hip/hip_runtime.h>

typedef _Float16 f16;
typedef __attribute__((ext_vector_type(16))) _Float16 v16h;
typedef __attribute__((ext_vector_type(8)))  float    v8f;
typedef __attribute__((ext_vector_type(4)))  float    v4f;

#define BB  4
#define NE  128
#define NV  1024
#define DIN 64
#define DD  128
#define TT  2

// ---------------------------------------------------------------------------
// WMMA GEMM: C[b] = act( A[b](MxK,f32) @ B[b](KxN) + bias + rowAdd[b] )
// B is supplied pre-transposed+converted: Bt[b] is (N,K) in f16.
// One wave computes a 16x64 tile (4 WMMA accumulators, A fragment reused).
// ---------------------------------------------------------------------------
__global__ __launch_bounds__(256)
void gemm_wmma(const float* __restrict__ A, const f16* __restrict__ Bt,
               float* __restrict__ C, const float* __restrict__ bias,
               const float* __restrict__ rowAdd, int batch, int M, int N, int K,
               long sA, long sB, long sC, int act)
{
    const int lane = threadIdx.x & 31;
    const int wid  = (int)((blockIdx.x * blockDim.x + threadIdx.x) >> 5);
    const int ntiles = N >> 6;           // 64-col tiles
    const int mtiles = M >> 4;           // 16-row tiles
    const int tpb = mtiles * ntiles;
    const int b = wid / tpb;
    if (b >= batch) return;
    const int t  = wid % tpb;
    const int m0 = (t / ntiles) << 4;
    const int n0 = (t % ntiles) << 6;

    const float* Ab = A + (long)b * sA;
    const f16*   Bb = Bt + (long)b * sB;
    float*       Cb = C + (long)b * sC;

    const int row  = lane & 15;
    const int half = lane >> 4;

    v8f acc[4] = {};
    const float* arow = Ab + (long)(m0 + row) * K;

    for (int kb = 0; kb < K; kb += 32) {
        // A fragment (contiguous k order, same as B): lane half h holds K = kb+16h..+15
        const float* ap = arow + kb + half * 16;
        v4f a0 = *(const v4f*)(ap + 0);
        v4f a1 = *(const v4f*)(ap + 4);
        v4f a2 = *(const v4f*)(ap + 8);
        v4f a3 = *(const v4f*)(ap + 12);
        v16h afrag;
#pragma unroll
        for (int j = 0; j < 4; ++j) {
            afrag[j]      = (f16)a0[j];
            afrag[4 + j]  = (f16)a1[j];
            afrag[8 + j]  = (f16)a2[j];
            afrag[12 + j] = (f16)a3[j];
        }
#pragma unroll
        for (int sub = 0; sub < 4; ++sub) {
            // B fragment: 16 consecutive K at kb + half*16 for column n0+sub*16+row
            const f16* bp = Bb + (long)(n0 + sub * 16 + row) * K + kb + half * 16;
            v16h bfrag = *(const v16h*)bp;
            acc[sub] = __builtin_amdgcn_wmma_f32_16x16x32_f16(
                false, afrag, false, bfrag, (short)0, acc[sub], false, false);
        }
    }
#pragma unroll
    for (int sub = 0; sub < 4; ++sub) {
        int n = n0 + sub * 16 + row;
        float add = 0.f;
        if (bias)   add += bias[n];
        if (rowAdd) add += rowAdd[b * N + n];
#pragma unroll
        for (int r = 0; r < 8; ++r) {
            int m = m0 + r + half * 8;
            float v = acc[sub][r] + add;
            if (act) v = fmaxf(v, 0.f);
            Cb[(long)m * N + n] = v;
        }
    }
}

// ---------------------------------------------------------------------------
// Transpose + convert f32 (K,N) -> f16 (N,K), optional elementwise sum of two
// sources (used to fold DeepSet twin weights w1+w2). LDS 32x33 tiles.
// ---------------------------------------------------------------------------
__global__ __launch_bounds__(256)
void tcvt_kernel(const float* __restrict__ s1, const float* __restrict__ s2,
                 f16* __restrict__ dst, int K, int N, int batch, long ss, long ds)
{
    __shared__ float tile[32][33];
    int kt = K >> 5, nt = N >> 5;
    int per = kt * nt;
    int b = blockIdx.x / per;
    if (b >= batch) return;
    int rem = blockIdx.x % per;
    int k0 = (rem / nt) << 5, n0 = (rem % nt) << 5;
    const float* S1 = s1 + (long)b * ss;
    const float* S2 = s2 ? s2 + (long)b * ss : nullptr;
    int tx = threadIdx.x & 31, ty = threadIdx.x >> 5;
#pragma unroll
    for (int i = 0; i < 32; i += 8) {
        long off = (long)(k0 + ty + i) * N + n0 + tx;
        float v = S1[off];
        if (S2) v += S2[off];
        tile[ty + i][tx] = v;
    }
    __syncthreads();
    f16* Dd = dst + (long)b * ds;
#pragma unroll
    for (int i = 0; i < 32; i += 8)
        Dd[(long)(n0 + ty + i) * K + k0 + tx] = (f16)tile[tx][ty + i];
}

// Plain f32 transpose (R,C) -> (C,R), batched (used for im^T).
__global__ __launch_bounds__(256)
void ftran_kernel(const float* __restrict__ src, float* __restrict__ dst,
                  int R, int Cc, int batch)
{
    __shared__ float tile[32][33];
    int rt = R >> 5, ct = Cc >> 5;
    int per = rt * ct;
    int b = blockIdx.x / per;
    if (b >= batch) return;
    int rem = blockIdx.x % per;
    int r0 = (rem / ct) << 5, c0 = (rem % ct) << 5;
    const float* S = src + (long)b * R * Cc;
    float*       Dd = dst + (long)b * R * Cc;
    int tx = threadIdx.x & 31, ty = threadIdx.x >> 5;
#pragma unroll
    for (int i = 0; i < 32; i += 8)
        tile[ty + i][tx] = S[(long)(r0 + ty + i) * Cc + c0 + tx];
    __syncthreads();
#pragma unroll
    for (int i = 0; i < 32; i += 8)
        Dd[(long)(c0 + ty + i) * R + r0 + tx] = tile[tx][ty + i];
}

__global__ void vadd_kernel(const float* a, const float* b, float* o, int n)
{
    int i = blockIdx.x * blockDim.x + threadIdx.x;
    if (i < n) o[i] = a[i] + b[i];
}

// ---------------------------------------------------------------------------
// Fused incidence kernel: i'[b,e,n] = sigmoid( sum_d relu(ve+ee+i*wi+bi)*wo + bo )
// Never materializes the (b,e,n,d) tensor. ve tile (64 rows) staged in LDS,
// reused across a chunk of 16 edges. Also produces im = i' * e_ind and writes
// the preds slot for this refinement step.
// ---------------------------------------------------------------------------
__global__ __launch_bounds__(64)
void incidence_kernel(const float* __restrict__ ve, const float* __restrict__ ee,
                      const float* __restrict__ e_ind,
                      const float* __restrict__ wi, const float* __restrict__ bi,
                      const float* __restrict__ wo, const float* __restrict__ bo,
                      float* __restrict__ i_t, float* __restrict__ im,
                      float* __restrict__ preds)
{
    __shared__ float ve_s[64][DD + 1];
    __shared__ float ee_s[DD];
    __shared__ float wi_s[DD], bi_s[DD], wo_s[DD];
    const int ntiles = NV / 64;   // 16
    const int ech    = NE / 16;   // 8
    int b   = blockIdx.x / (ntiles * ech);
    int rem = blockIdx.x % (ntiles * ech);
    int n0  = (rem / ech) * 64;
    int e0  = (rem % ech) * 16;
    int tid = threadIdx.x;

    for (int i = tid; i < DD; i += 64) { wi_s[i] = wi[i]; bi_s[i] = bi[i]; wo_s[i] = wo[i]; }
    const float* vp = ve + ((long)b * NV + n0) * DD;
    for (int i = tid; i < 64 * DD; i += 64) ve_s[i >> 7][i & (DD - 1)] = vp[i];
    __syncthreads();

    float bo_v = bo[0];
    int n = n0 + tid;
    for (int ec = 0; ec < 16; ++ec) {
        int e = e0 + ec;
        const float* ep = ee + ((long)b * NE + e) * DD;
        for (int i = tid; i < DD; i += 64) ee_s[i] = ep[i];
        __syncthreads();
        long idx = ((long)b * NE + e) * NV + n;
        float iv = i_t[idx];
        float acc = 0.f;
#pragma unroll 4
        for (int d = 0; d < DD; ++d) {
            float h = ve_s[tid][d] + ee_s[d] + fmaf(iv, wi_s[d], bi_s[d]);
            acc = fmaf(fmaxf(h, 0.f), wo_s[d], acc);
        }
        float inew = 1.f / (1.f + __expf(-(acc + bo_v)));
        i_t[idx] = inew;
        im[idx]  = inew * e_ind[b * NE + e];
        preds[((long)b * NE + e) * (NV + 1) + n] = inew;
        __syncthreads();
    }
}

// e_ind = sigmoid(e_t @ ind_w + ind_b); one wave per edge row; also writes
// preds[...,1024].
__global__ __launch_bounds__(256)
void eind_kernel(const float* __restrict__ e_t, const float* __restrict__ ind_w,
                 const float* __restrict__ ind_b, float* __restrict__ e_ind,
                 float* __restrict__ preds, int rows)
{
    int wid  = (blockIdx.x * blockDim.x + threadIdx.x) >> 5;
    int lane = threadIdx.x & 31;
    if (wid >= rows) return;
    const float* x = e_t + (long)wid * DD;
    float s = 0.f;
#pragma unroll
    for (int i = 0; i < 4; ++i) s += x[lane + 32 * i] * ind_w[lane + 32 * i];
#pragma unroll
    for (int m = 16; m; m >>= 1) s += __shfl_xor(s, m, 32);
    if (lane == 0) {
        float v = 1.f / (1.f + __expf(-(s + ind_b[0])));
        e_ind[wid] = v;
        preds[(long)wid * (NV + 1) + NV] = v;
    }
}

// LayerNorm over a row of width W (multiple of 32, <=384); one wave per row.
__global__ __launch_bounds__(256)
void ln_rows_kernel(const float* __restrict__ in, float* __restrict__ out,
                    const float* __restrict__ g, const float* __restrict__ beta,
                    int rows, int W)
{
    int wid  = (blockIdx.x * blockDim.x + threadIdx.x) >> 5;
    int lane = threadIdx.x & 31;
    if (wid >= rows) return;
    const float* x = in + (long)wid * W;
    int per = W >> 5;
    float vals[12];
    float s = 0.f;
    for (int i = 0; i < per; ++i) { vals[i] = x[lane + 32 * i]; s += vals[i]; }
    for (int m = 16; m; m >>= 1) s += __shfl_xor(s, m, 32);
    float mean = s / (float)W;
    float var = 0.f;
    for (int i = 0; i < per; ++i) { float d = vals[i] - mean; var += d * d; }
    for (int m = 16; m; m >>= 1) var += __shfl_xor(var, m, 32);
    float rstd = rsqrtf(var / (float)W + 1e-5f);
    float* o = out + (long)wid * W;
    for (int i = 0; i < per; ++i) {
        int c = lane + 32 * i;
        o[c] = (vals[i] - mean) * rstd * g[c] + beta[c];
    }
}

// LN(concat(parts x 128-wide tensors)); one wave per row.
__global__ __launch_bounds__(256)
void concat_ln_kernel(const float* __restrict__ p0, const float* __restrict__ p1,
                      const float* __restrict__ p2, const float* __restrict__ g,
                      const float* __restrict__ beta, float* __restrict__ out,
                      int rows, int parts)
{
    int wid  = (blockIdx.x * blockDim.x + threadIdx.x) >> 5;
    int lane = threadIdx.x & 31;
    if (wid >= rows) return;
    const float* ps[3] = { p0, p1, p2 };
    int W = parts * DD;
    float vals[12];
    float s = 0.f;
    for (int p = 0; p < parts; ++p) {
        const float* x = ps[p] + (long)wid * DD;
#pragma unroll
        for (int i = 0; i < 4; ++i) { float v = x[lane + 32 * i]; vals[p * 4 + i] = v; s += v; }
    }
    for (int m = 16; m; m >>= 1) s += __shfl_xor(s, m, 32);
    float mean = s / (float)W;
    float var = 0.f;
    for (int i = 0; i < parts * 4; ++i) { float d = vals[i] - mean; var += d * d; }
    for (int m = 16; m; m >>= 1) var += __shfl_xor(var, m, 32);
    float rstd = rsqrtf(var / (float)W + 1e-5f);
    float* o = out + (long)wid * W;
    for (int p = 0; p < parts; ++p)
        for (int i = 0; i < 4; ++i) {
            int c = p * DD + lane + 32 * i;
            o[c] = (vals[p * 4 + i] - mean) * rstd * g[c] + beta[c];
        }
}

// xio = LN(xio + y)  (in-place, row-private so deterministic on replay)
__global__ __launch_bounds__(256)
void residual_ln_kernel(float* __restrict__ xio, const float* __restrict__ y,
                        const float* __restrict__ g, const float* __restrict__ beta,
                        int rows)
{
    int wid  = (blockIdx.x * blockDim.x + threadIdx.x) >> 5;
    int lane = threadIdx.x & 31;
    if (wid >= rows) return;
    float vals[4];
    float s = 0.f;
#pragma unroll
    for (int i = 0; i < 4; ++i) {
        int c = lane + 32 * i;
        vals[i] = xio[(long)wid * DD + c] + y[(long)wid * DD + c];
        s += vals[i];
    }
    for (int m = 16; m; m >>= 1) s += __shfl_xor(s, m, 32);
    float mean = s / (float)DD;
    float var = 0.f;
#pragma unroll
    for (int i = 0; i < 4; ++i) { float d = vals[i] - mean; var += d * d; }
    for (int m = 16; m; m >>= 1) var += __shfl_xor(var, m, 32);
    float rstd = rsqrtf(var / (float)DD + 1e-5f);
#pragma unroll
    for (int i = 0; i < 4; ++i) {
        int c = lane + 32 * i;
        xio[(long)wid * DD + c] = (vals[i] - mean) * rstd * g[c] + beta[c];
    }
}

// Column mean over the set dimension: out[b][w] = mean_r in[b][r][w]
__global__ void colmean_kernel(const float* __restrict__ in, float* __restrict__ out,
                               int rows, int W, int batch)
{
    int idx = blockIdx.x * blockDim.x + threadIdx.x;
    if (idx >= batch * W) return;
    int b = idx / W, w = idx % W;
    const float* p = in + (long)b * rows * W + w;
    float s = 0.f;
    for (int r = 0; r < rows; ++r) s += p[(long)r * W];
    out[idx] = s / (float)rows;
}

// out[b][n] = -sum_k mean[b][k] * w2[k][n]  (rank-1 mean correction term)
__global__ void negmatvec_kernel(const float* __restrict__ mean, const float* __restrict__ w2,
                                 float* __restrict__ out, int K, int N, int batch)
{
    int idx = blockIdx.x * blockDim.x + threadIdx.x;
    if (idx >= batch * N) return;
    int b = idx / N, n = idx % N;
    const float* m = mean + (long)b * K;
    float s = 0.f;
    for (int k = 0; k < K; ++k) s += m[k] * w2[(long)k * N + n];
    out[idx] = -s;
}

// ---------------------------------------------------------------------------
extern "C" void kernel_launch(void* const* d_in, const int* in_sizes, int n_in,
                              void* d_out, int out_size, void* d_ws, size_t ws_size,
                              hipStream_t stream)
{
    (void)in_sizes; (void)n_in; (void)out_size; (void)ws_size;

    const float* in_inputs = (const float*)d_in[0];
    const float* in_e   = (const float*)d_in[1];
    const float* in_v   = (const float*)d_in[2];
    const float* in_i   = (const float*)d_in[3];
    const float* pw     = (const float*)d_in[4];
    const float* inc_we = (const float*)d_in[5];
    const float* inc_wn = (const float*)d_in[6];
    const float* inc_wi = (const float*)d_in[7];
    const float* inc_wo = (const float*)d_in[8];
    const float* ind_w  = (const float*)d_in[9];
    const float* me_w1a = (const float*)d_in[10];
    const float* me_w2a = (const float*)d_in[11];
    const float* me_w1b = (const float*)d_in[12];
    const float* me_w2b = (const float*)d_in[13];
    const float* mn_w1a = (const float*)d_in[14];
    const float* mn_w2a = (const float*)d_in[15];
    const float* mn_w1b = (const float*)d_in[16];
    const float* mn_w2b = (const float*)d_in[17];
    const float* pb     = (const float*)d_in[18];
    const float* inc_be = (const float*)d_in[19];
    const float* inc_bn = (const float*)d_in[20];
    const float* inc_bi = (const float*)d_in[21];
    const float* inc_bo = (const float*)d_in[22];
    const float* ind_b  = (const float*)d_in[23];
    const float* me_b1a = (const float*)d_in[24];
    const float* me_b2a = (const float*)d_in[25];
    const float* me_b1b = (const float*)d_in[26];
    const float* me_b2b = (const float*)d_in[27];
    const float* mn_b1a = (const float*)d_in[28];
    const float* mn_b2a = (const float*)d_in[29];
    const float* mn_b1b = (const float*)d_in[30];
    const float* mn_b2b = (const float*)d_in[31];
    const float* me_g   = (const float*)d_in[32];
    const float* mn_g   = (const float*)d_in[33];
    const float* pre_e_g= (const float*)d_in[34];
    const float* pre_n_g= (const float*)d_in[35];
    const float* ln_e_g = (const float*)d_in[36];
    const float* ln_n_g = (const float*)d_in[37];
    const float* me_bg  = (const float*)d_in[38];
    const float* mn_bg  = (const float*)d_in[39];
    const float* pre_e_b= (const float*)d_in[40];
    const float* pre_n_b= (const float*)d_in[41];
    const float* ln_e_b = (const float*)d_in[42];
    const float* ln_n_b = (const float*)d_in[43];

    // ---- workspace carve-out (256B aligned) ----
    char* ws = (char*)d_ws;
    size_t cur = 0;
    auto alloc = [&](size_t bytes) -> void* {
        void* p = ws + cur;
        cur += (bytes + 255) & ~(size_t)255;
        return p;
    };
    f16* pwT    = (f16*)alloc(sizeof(f16) * DD * DIN);
    f16* weT    = (f16*)alloc(sizeof(f16) * DD * DD);
    f16* wnT    = (f16*)alloc(sizeof(f16) * DD * DD);
    f16* mewsaT = (f16*)alloc(sizeof(f16) * DD * 2 * DD);
    f16* mewsbT = (f16*)alloc(sizeof(f16) * DD * DD);
    f16* mnwsaT = (f16*)alloc(sizeof(f16) * DD * 3 * DD);
    f16* mnwsbT = (f16*)alloc(sizeof(f16) * DD * DD);
    f16* vtT    = (f16*)alloc(sizeof(f16) * (size_t)BB * DD * NV);
    f16* etT    = (f16*)alloc(sizeof(f16) * (size_t)BB * DD * DD);
    float* me_bsa = (float*)alloc(4 * DD);
    float* me_bsb = (float*)alloc(4 * DD);
    float* mn_bsa = (float*)alloc(4 * DD);
    float* mn_bsb = (float*)alloc(4 * DD);
    float* x      = (float*)alloc(4L * BB * NV * DD);
    float* ve     = (float*)alloc(4L * BB * NV * DD);
    float* ee     = (float*)alloc(4L * BB * NE * DD);
    float* e_ind  = (float*)alloc(4L * BB * NE);
    float* e_cur  = (float*)alloc(4L * BB * NE * DD);
    float* v_cur  = (float*)alloc(4L * BB * NV * DD);
    float* i_cur  = (float*)alloc(4L * BB * NE * NV);
    float* im     = (float*)alloc(4L * BB * NE * NV);
    float* imT    = (float*)alloc(4L * BB * NE * NV);
    float* upd_e  = (float*)alloc(4L * BB * NE * DD);
    float* ce     = (float*)alloc(4L * BB * NE * 2 * DD);
    float* mean_e = (float*)alloc(4L * BB * 2 * DD);
    float* nmt_ae = (float*)alloc(4L * BB * DD);
    float* t1e    = (float*)alloc(4L * BB * NE * DD);
    float* u_e    = (float*)alloc(4L * BB * NE * DD);
    float* mean_ue= (float*)alloc(4L * BB * DD);
    float* nmt_be = (float*)alloc(4L * BB * DD);
    float* out2e  = (float*)alloc(4L * BB * NE * DD);
    float* upd_n  = (float*)alloc(4L * BB * NV * DD);
    float* cn     = (float*)alloc(4L * BB * NV * 3 * DD);
    float* mean_n = (float*)alloc(4L * BB * 3 * DD);
    float* nmt_an = (float*)alloc(4L * BB * DD);
    float* t1n    = (float*)alloc(4L * BB * NV * DD);
    float* u_n    = (float*)alloc(4L * BB * NV * DD);
    float* mean_un= (float*)alloc(4L * BB * DD);
    float* nmt_bn = (float*)alloc(4L * BB * DD);
    float* out2n  = (float*)alloc(4L * BB * NV * DD);

    auto gemm = [&](const float* A, const f16* Bt, float* C, const float* bias,
                    const float* rowAdd, int batch, int M, int N, int K,
                    long sA, long sB, long sC, int act) {
        int waves = batch * (M / 16) * (N / 64);
        gemm_wmma<<<(waves * 32 + 255) / 256, 256, 0, stream>>>(
            A, Bt, C, bias, rowAdd, batch, M, N, K, sA, sB, sC, act);
    };
    auto tcvt = [&](const float* s1, const float* s2, f16* dst, int K, int N,
                    int batch, long ss, long ds) {
        tcvt_kernel<<<batch * (K / 32) * (N / 32), 256, 0, stream>>>(
            s1, s2, dst, K, N, batch, ss, ds);
    };
    auto wg = [](int waves) { return (waves * 32 + 255) / 256; };

    // ---- state init (do not mutate inputs) ----
    hipMemcpyAsync(e_cur, in_e, 4L * BB * NE * DD, hipMemcpyDeviceToDevice, stream);
    hipMemcpyAsync(v_cur, in_v, 4L * BB * NV * DD, hipMemcpyDeviceToDevice, stream);
    hipMemcpyAsync(i_cur, in_i, 4L * BB * NE * NV, hipMemcpyDeviceToDevice, stream);

    // ---- weight prep: transpose+cvt to f16 (N,K); fold DeepSet twin weights ----
    tcvt(pw,     nullptr, pwT,    DIN,    DD, 1, 0, 0);
    tcvt(inc_we, nullptr, weT,    DD,     DD, 1, 0, 0);
    tcvt(inc_wn, nullptr, wnT,    DD,     DD, 1, 0, 0);
    tcvt(me_w1a, me_w2a,  mewsaT, 2 * DD, DD, 1, 0, 0);
    tcvt(me_w1b, me_w2b,  mewsbT, DD,     DD, 1, 0, 0);
    tcvt(mn_w1a, mn_w2a,  mnwsaT, 3 * DD, DD, 1, 0, 0);
    tcvt(mn_w1b, mn_w2b,  mnwsbT, DD,     DD, 1, 0, 0);
    vadd_kernel<<<1, 128, 0, stream>>>(me_b1a, me_b2a, me_bsa, DD);
    vadd_kernel<<<1, 128, 0, stream>>>(me_b1b, me_b2b, me_bsb, DD);
    vadd_kernel<<<1, 128, 0, stream>>>(mn_b1a, mn_b2a, mn_bsa, DD);
    vadd_kernel<<<1, 128, 0, stream>>>(mn_b1b, mn_b2b, mn_bsb, DD);

    // x = inputs @ pw + pb
    gemm(in_inputs, pwT, x, pb, nullptr, 1, BB * NV, DD, DIN, 0, 0, 0, 0);

    float* outp = (float*)d_out;
    for (int t = 0; t < TT; ++t) {
        float* preds_t = outp + (long)t * BB * NE * (NV + 1);

        // incidence MLP inputs
        gemm(v_cur, wnT, ve, inc_bn, nullptr, 1, BB * NV, DD, DD, 0, 0, 0, 0);
        gemm(e_cur, weT, ee, inc_be, nullptr, 1, BB * NE, DD, DD, 0, 0, 0, 0);
        eind_kernel<<<wg(BB * NE), 256, 0, stream>>>(e_cur, ind_w, ind_b, e_ind,
                                                     preds_t, BB * NE);
        incidence_kernel<<<BB * (NV / 64) * (NE / 16), 64, 0, stream>>>(
            ve, ee, e_ind, inc_wi, inc_bi, inc_wo, inc_bo, i_cur, im, preds_t);

        // ---- edge update ----
        tcvt(v_cur, nullptr, vtT, NV, DD, BB, (long)NV * DD, (long)DD * NV);
        gemm(im, vtT, upd_e, nullptr, nullptr, BB, NE, DD, NV,
             (long)NE * NV, (long)DD * NV, (long)NE * DD, 0);
        concat_ln_kernel<<<wg(BB * NE), 256, 0, stream>>>(
            e_cur, upd_e, nullptr, pre_e_g, pre_e_b, ce, BB * NE, 2);
        colmean_kernel<<<(BB * 2 * DD + 255) / 256, 256, 0, stream>>>(ce, mean_e, NE, 2 * DD, BB);
        negmatvec_kernel<<<(BB * DD + 255) / 256, 256, 0, stream>>>(mean_e, me_w2a, nmt_ae, 2 * DD, DD, BB);
        gemm(ce, mewsaT, t1e, me_bsa, nmt_ae, BB, NE, DD, 2 * DD,
             (long)NE * 2 * DD, 0, (long)NE * DD, 1);
        ln_rows_kernel<<<wg(BB * NE), 256, 0, stream>>>(t1e, u_e, me_g, me_bg, BB * NE, DD);
        colmean_kernel<<<(BB * DD + 255) / 256, 256, 0, stream>>>(u_e, mean_ue, NE, DD, BB);
        negmatvec_kernel<<<(BB * DD + 255) / 256, 256, 0, stream>>>(mean_ue, me_w2b, nmt_be, DD, DD, BB);
        gemm(u_e, mewsbT, out2e, me_bsb, nmt_be, BB, NE, DD, DD,
             (long)NE * DD, 0, (long)NE * DD, 0);
        residual_ln_kernel<<<wg(BB * NE), 256, 0, stream>>>(e_cur, out2e, ln_e_g, ln_e_b, BB * NE);

        // ---- node update ----
        tcvt(e_cur, nullptr, etT, DD, DD, BB, (long)DD * DD, (long)DD * DD);
        ftran_kernel<<<BB * (NE / 32) * (NV / 32), 256, 0, stream>>>(im, imT, NE, NV, BB);
        gemm(imT, etT, upd_n, nullptr, nullptr, BB, NV, DD, NE,
             (long)NV * NE, (long)DD * DD, (long)NV * DD, 0);
        concat_ln_kernel<<<wg(BB * NV), 256, 0, stream>>>(
            x, v_cur, upd_n, pre_n_g, pre_n_b, cn, BB * NV, 3);
        colmean_kernel<<<(BB * 3 * DD + 255) / 256, 256, 0, stream>>>(cn, mean_n, NV, 3 * DD, BB);
        negmatvec_kernel<<<(BB * DD + 255) / 256, 256, 0, stream>>>(mean_n, mn_w2a, nmt_an, 3 * DD, DD, BB);
        gemm(cn, mnwsaT, t1n, mn_bsa, nmt_an, BB, NV, DD, 3 * DD,
             (long)NV * 3 * DD, 0, (long)NV * DD, 1);
        ln_rows_kernel<<<wg(BB * NV), 256, 0, stream>>>(t1n, u_n, mn_g, mn_bg, BB * NV, DD);
        colmean_kernel<<<(BB * DD + 255) / 256, 256, 0, stream>>>(u_n, mean_un, NV, DD, BB);
        negmatvec_kernel<<<(BB * DD + 255) / 256, 256, 0, stream>>>(mean_un, mn_w2b, nmt_bn, DD, DD, BB);
        gemm(u_n, mnwsbT, out2n, mn_bsb, nmt_bn, BB, NV, DD, DD,
             (long)NV * DD, 0, (long)NV * DD, 0);
        residual_ln_kernel<<<wg(BB * NV), 256, 0, stream>>>(v_cur, out2n, ln_n_g, ln_n_b, BB * NV);
    }

    // ---- final outputs: (preds already written) e_t, v_t, i_t ----
    long predsN = (long)TT * BB * NE * (NV + 1);
    hipMemcpyAsync(outp + predsN, e_cur, 4L * BB * NE * DD, hipMemcpyDeviceToDevice, stream);
    hipMemcpyAsync(outp + predsN + (long)BB * NE * DD, v_cur, 4L * BB * NV * DD,
                   hipMemcpyDeviceToDevice, stream);
    hipMemcpyAsync(outp + predsN + (long)BB * NE * DD + (long)BB * NV * DD, i_cur,
                   4L * BB * NE * NV, hipMemcpyDeviceToDevice, stream);
}
